// Attention_5823975653620
// MI455X (gfx1250) — matrix operand order repaired and checked by reading, not compiled, not run on hardware
//
#include <hip/hip_runtime.h>
#include <hip/hip_bf16.h>

// ---------------------------------------------------------------------------
// MHA block for B=4, N=1024, C=768, H=24, D=32 on gfx1250 (wave32, WMMA).
// All matmuls use v_wmma_f32_16x16x32_bf16 (bf16 in, f32 accumulate).
// Data movement: GLOBAL_LOAD_ASYNC_TO_LDS (ASYNCcnt) for the GEMM A-tile,
// TDM tensor_load_to_lds (TENSORcnt) for the attention K-tile.
// ---------------------------------------------------------------------------

typedef __attribute__((ext_vector_type(16))) __bf16 v16bf;
typedef __attribute__((ext_vector_type(8)))  float  v8f;
typedef __attribute__((ext_vector_type(4)))  unsigned int u32x4;
typedef __attribute__((ext_vector_type(8)))  unsigned int u32x8;

#define ATT_B 4
#define ATT_N 1024
#define ATT_C 768
#define ATT_H 24
#define ATT_D 32
#define ATT_BH (ATT_B * ATT_H)           // 96
#define QKV_N (3 * ATT_C)                // 2304
#define ATT_SCALE 0.17677669529663687f   // 32^-0.5

__device__ __forceinline__ unsigned short f2bf(float f) {
  union { float f; unsigned int u; } x; x.f = f;
  unsigned int u = x.u;
  return (unsigned short)((u + 0x7FFFu + ((u >> 16) & 1u)) >> 16);
}

// --------------------------- f32 -> bf16 convert ---------------------------
__global__ __launch_bounds__(256) void cvt_bf16_kernel(
    const float* __restrict__ src, unsigned short* __restrict__ dst, int n) {
  int i = blockIdx.x * 256 + threadIdx.x;
  if (i < n) dst[i] = f2bf(src[i]);
}

// ------------------------------ WMMA GEMM ----------------------------------
// C[M,N] = A[M,K] * B[K,N], A/B bf16 row-major.
// Block tile 128x128, 8 waves (2 along M x 4 along N), each wave 64x32,
// K-step 32 (one 16x16x32 bf16 WMMA per fragment).
// mode 0: store f32 to Cf.  mode 1: scatter bf16 qkv to [s][b*H+h][n][d],
//         folding ATT_SCALE into Q (s==0).
#define GTM 128
#define GTN 128
#define GTK 32

__global__ __launch_bounds__(256) void gemm_wmma_bf16_kernel(
    const unsigned short* __restrict__ A, const unsigned short* __restrict__ Bm,
    float* __restrict__ Cf, unsigned short* __restrict__ Cq,
    int M, int N, int K, int mode) {
  __shared__ __align__(16) unsigned short As[GTM][GTK];   // 128x32 bf16 = 8KB
  __shared__ __align__(16) unsigned short Bt[GTN][GTK];   // B transposed, 8KB

  const int tid = threadIdx.x;
  const int wid = tid >> 5;
  const int lane = tid & 31;
  const int halfg = lane >> 4;   // 0: lanes 0-15, 1: lanes 16-31
  const int l16 = lane & 15;
  const int bm = blockIdx.x;     // M / 128
  const int bn = blockIdx.y;     // N / 128
  const int wm = wid & 1;        // wave M slot (0..1) -> 64 rows
  const int wn = wid >> 1;       // wave N slot (0..3) -> 32 cols

  v8f acc[4][2];
  const v8f vzero = {0.f, 0.f, 0.f, 0.f, 0.f, 0.f, 0.f, 0.f};
#pragma unroll
  for (int fm = 0; fm < 4; ++fm)
#pragma unroll
    for (int fn = 0; fn < 2; ++fn) acc[fm][fn] = vzero;

  // per-thread async A-tile slot: 32 contiguous bytes
  const int arow = tid >> 1;
  const int acol = (tid & 1) * 16;
  const unsigned int lds_a = (unsigned int)(size_t)(&As[arow][acol]);

  for (int k0 = 0; k0 < K; k0 += GTK) {
    // ---- A tile (128x32) via async global->LDS (ASYNCcnt), 2x b128/thread
    {
      const unsigned short* src = A + (size_t)(bm * GTM + arow) * K + k0 + acol;
      unsigned long long ga = (unsigned long long)(size_t)src;
      asm volatile("global_load_async_to_lds_b128 %0, %1, off"
                   :: "v"(lds_a), "v"(ga) : "memory");
      asm volatile("global_load_async_to_lds_b128 %0, %1, off offset:16"
                   :: "v"(lds_a), "v"(ga) : "memory");
      if (k0 + GTK < K) __builtin_prefetch(src + GTK, 0, 0);  // global_prefetch_b8
    }
    // ---- B tile transposed (needs reshape -> manual): Bt[n][k] = B[k][n]
    {
      int k = tid >> 3;
      int nb = (tid & 7) * 16;
      const unsigned short* src = Bm + (size_t)(k0 + k) * N + bn * GTN + nb;
      union { uint4 q[2]; unsigned short s[16]; } u;
      u.q[0] = *(const uint4*)(src);
      u.q[1] = *(const uint4*)(src + 8);
#pragma unroll
      for (int j = 0; j < 16; ++j) Bt[nb + j][k] = u.s[j];
      if (k0 + GTK < K) __builtin_prefetch(src + (size_t)GTK * N, 0, 0);
    }
    asm volatile("s_wait_asynccnt 0x0" ::: "memory");
    __syncthreads();

    // ---- A fragments (16x32 bf16, contiguous k per lane half, same as B):
    // lane: m = lane%16; VGPR i : k = (lane/16)*16 + 2i (dwords halfg*8+i)
    union FragAB { unsigned int u[8]; v16bf v; };
    FragAB fa[4];
#pragma unroll
    for (int fm = 0; fm < 4; ++fm) {
      int m = wm * 64 + fm * 16 + l16;
      const unsigned int* rowp = (const unsigned int*)(&As[m][0]);
#pragma unroll
      for (int i = 0; i < 8; ++i) fa[fm].u[i] = rowp[halfg * 8 + i];
    }
    // ---- B fragments (32x16 bf16): lane: n = lane%16; k = (lane/16)*16 + 2i
    //      -> contiguous dwords halfg*8 + i from Bt[n][*]
    FragAB fb[2];
#pragma unroll
    for (int fn = 0; fn < 2; ++fn) {
      int n = wn * 32 + fn * 16 + l16;
      const unsigned int* colp = (const unsigned int*)(&Bt[n][0]);
#pragma unroll
      for (int i = 0; i < 8; ++i) fb[fn].u[i] = colp[halfg * 8 + i];
    }
#pragma unroll
    for (int fm = 0; fm < 4; ++fm)
#pragma unroll
      for (int fn = 0; fn < 2; ++fn)
        acc[fm][fn] = __builtin_amdgcn_wmma_f32_16x16x32_bf16(
            false, fa[fm].v, false, fb[fn].v, (short)0, acc[fm][fn],
            false, false);
    __syncthreads();
  }

  // ---- epilogue: C layout: VGPR r holds (m = r + 8*(lane/16), n = lane%16)
#pragma unroll
  for (int fm = 0; fm < 4; ++fm) {
#pragma unroll
    for (int fn = 0; fn < 2; ++fn) {
#pragma unroll
      for (int r = 0; r < 8; ++r) {
        int i = bm * GTM + wm * 64 + fm * 16 + r + halfg * 8;
        int j = bn * GTN + wn * 32 + fn * 16 + l16;
        float v = acc[fm][fn][r];
        if (mode == 0) {
          Cf[(size_t)i * N + j] = v;
        } else {
          int s = j / ATT_C;
          int rem = j - s * ATT_C;
          int h = rem >> 5, d = rem & 31;
          int b = i >> 10, nn = i & 1023;
          if (s == 0) v *= ATT_SCALE;  // fold softmax scale into Q
          size_t idx =
              (((size_t)s * ATT_BH + (size_t)b * ATT_H + h) * ATT_N + nn) *
                  ATT_D + d;
          Cq[idx] = f2bf(v);
        }
      }
    }
  }
}

// ------------------------- flash-style attention ---------------------------
// Grid: (96 head-batches, 8 query tiles of 128). 8 waves; wave owns 16 q rows.
// K-tile arrives via TDM tensor_load_to_lds (one DMA per 32-key block,
// issued by wave 0, waited with s_wait_tensorcnt). V-tile is transposed
// manually. S = Q(16x32) x K^T via 2 WMMAs, online softmax, P re-laid out
// via per-wave LDS buffer, O += P x V via 2 WMMAs.
__global__ __launch_bounds__(256) void attn_wmma_kernel(
    const unsigned short* __restrict__ Q, const unsigned short* __restrict__ Km,
    const unsigned short* __restrict__ Vm, unsigned short* __restrict__ AO) {
  __shared__ __align__(16) unsigned short Ks[32][32];      // [key][d]   2KB
  __shared__ __align__(16) unsigned short Vt[32][32];      // [d][key]   2KB
  __shared__ __align__(16) unsigned short Ps[8][16][32];   // per-wave P 8KB

  const int tid = threadIdx.x;
  const int wid = tid >> 5;
  const int lane = tid & 31;
  const int halfg = lane >> 4;
  const int l16 = lane & 15;
  const int bh = blockIdx.x;  // 0..95
  const int qt = blockIdx.y;  // 0..7

  const unsigned short* Qb = Q + (size_t)bh * ATT_N * ATT_D;
  const unsigned short* Kb = Km + (size_t)bh * ATT_N * ATT_D;
  const unsigned short* Vb = Vm + (size_t)bh * ATT_N * ATT_D;

  union FragAB { unsigned int u[8]; v16bf v; };

  // Persistent Q fragment: wave rows qt*128 + wid*16 + (0..15), full D=32.
  FragAB fq;
  {
    int qrow = qt * 128 + wid * 16 + l16;
    const unsigned int* qp = (const unsigned int*)(Qb + (size_t)qrow * ATT_D);
#pragma unroll
    for (int i = 0; i < 8; ++i) fq.u[i] = qp[halfg * 8 + i];
  }

  const v8f vzero = {0.f, 0.f, 0.f, 0.f, 0.f, 0.f, 0.f, 0.f};
  v8f O0 = vzero, O1 = vzero;
  float mrow[8], lrow[8];
#pragma unroll
  for (int r = 0; r < 8; ++r) { mrow[r] = -1e30f; lrow[r] = 0.f; }

  const unsigned int ldsK = (unsigned int)(size_t)(&Ks[0][0]);

  for (int kb = 0; kb < ATT_N / 32; ++kb) {
    // ---- K tile (32x32 bf16, row-major) via TDM, issued by wave 0 only.
    if (tid < 32) {
      unsigned long long ga =
          (unsigned long long)(size_t)(Kb + (size_t)kb * 32 * ATT_D);
      // D# group 0: count=1 | lds_addr | global_addr[56:0] | type=2
      u32x4 g0;
      g0.x = 1u;                                   // count=1, no gather
      g0.y = ldsK;                                 // lds_addr
      g0.z = (unsigned int)(ga & 0xFFFFFFFFu);     // global_addr[31:0]
      g0.w = (unsigned int)((ga >> 32) & 0x01FFFFFFu) | 0x80000000u; // type=2
      // D# group 1: data_size=2B, tensor_dim0=32, dim0_stride=32,
      //             tensor_dim1=1024, tile 32x32.
      u32x8 g1;
      g1.s0 = 0x00010000u;        // workgroup_mask=0, data_size=1 (2 bytes)
      g1.s1 = (32u << 16);        // tensor_dim0[15:0]=32
      g1.s2 = (1024u << 16);      // tensor_dim0[31:16]=0, tensor_dim1[15:0]=1024
      g1.s3 = (32u << 16);        // tensor_dim1[31:16]=0, tile_dim0=32
      g1.s4 = 32u;                // tile_dim1=32, tile_dim2=0
      g1.s5 = 32u;                // tensor_dim0_stride[31:0]=32
      g1.s6 = 0u;                 // stride msbs / tensor_dim1_stride = 0
      g1.s7 = 0u;
      u32x4 gz = {0u, 0u, 0u, 0u};  // groups 2/3: dims 2..4 unused
      asm volatile("tensor_load_to_lds %0, %1, %2, %3"
                   :: "s"(g0), "s"(g1), "s"(gz), "s"(gz) : "memory");
      __builtin_amdgcn_s_wait_tensorcnt(0);
    }
    // ---- V tile transposed (needs reshape -> manual): Vt[d][key]
    {
      int key = tid >> 3;
      int db = (tid & 7) * 4;
      const unsigned short* vsrc = Vb + (size_t)(kb * 32 + key) * ATT_D + db;
#pragma unroll
      for (int j = 0; j < 4; ++j) Vt[db + j][key] = vsrc[j];
    }
    __syncthreads();

    // scores: B = K_sub^T (32x16): lane n = key, k = d -> Ks[key] contiguous
    FragAB fk0, fk1;
    {
      const unsigned int* p0 = (const unsigned int*)(&Ks[l16][0]);
      const unsigned int* p1 = (const unsigned int*)(&Ks[16 + l16][0]);
#pragma unroll
      for (int i = 0; i < 8; ++i) { fk0.u[i] = p0[halfg * 8 + i];
                                    fk1.u[i] = p1[halfg * 8 + i]; }
    }
    v8f S0 = __builtin_amdgcn_wmma_f32_16x16x32_bf16(
        false, fq.v, false, fk0.v, (short)0, vzero, false, false);
    v8f S1 = __builtin_amdgcn_wmma_f32_16x16x32_bf16(
        false, fq.v, false, fk1.v, (short)0, vzero, false, false);

    // online softmax (rows r + 8*halfg; 16-lane row reductions)
    float p0v[8], p1v[8];
#pragma unroll
    for (int r = 0; r < 8; ++r) {
      float mx = fmaxf(S0[r], S1[r]);
#pragma unroll
      for (int off = 8; off >= 1; off >>= 1)
        mx = fmaxf(mx, __shfl_xor(mx, off, 16));
      float mnew = fmaxf(mrow[r], mx);
      float corr = __expf(mrow[r] - mnew);
      mrow[r] = mnew;
      O0[r] *= corr;
      O1[r] *= corr;
      p0v[r] = __expf(S0[r] - mnew);
      p1v[r] = __expf(S1[r] - mnew);
      float sum = p0v[r] + p1v[r];
#pragma unroll
      for (int off = 8; off >= 1; off >>= 1) sum += __shfl_xor(sum, off, 16);
      lrow[r] = lrow[r] * corr + sum;
    }

    // bounce P through per-wave LDS to re-shape C-layout -> A-fragment layout
#pragma unroll
    for (int r = 0; r < 8; ++r) {
      int row = r + halfg * 8;
      Ps[wid][row][l16]      = f2bf(p0v[r]);
      Ps[wid][row][16 + l16] = f2bf(p1v[r]);
    }
    asm volatile("s_wait_dscnt 0" ::: "memory");
    __builtin_amdgcn_wave_barrier();

    FragAB fp;
    {
      const unsigned int* pp = (const unsigned int*)(&Ps[wid][l16][0]);
#pragma unroll
      for (int i = 0; i < 8; ++i) fp.u[i] = pp[halfg * 8 + i];
    }
    // PV: B = V_sub (32 keys x 16 dims): lane n = dim, k = key -> Vt[dim] row
    FragAB fv0, fv1;
    {
      const unsigned int* p0 = (const unsigned int*)(&Vt[l16][0]);
      const unsigned int* p1 = (const unsigned int*)(&Vt[16 + l16][0]);
#pragma unroll
      for (int i = 0; i < 8; ++i) { fv0.u[i] = p0[halfg * 8 + i];
                                    fv1.u[i] = p1[halfg * 8 + i]; }
    }
    O0 = __builtin_amdgcn_wmma_f32_16x16x32_bf16(
        false, fp.v, false, fv0.v, (short)0, O0, false, false);
    O1 = __builtin_amdgcn_wmma_f32_16x16x32_bf16(
        false, fp.v, false, fv1.v, (short)0, O1, false, false);
    __syncthreads();
  }

  // epilogue -> AO[b*N + q][h*32 + d] (bf16), normalized by l
  const int b = bh / ATT_H, h = bh % ATT_H;
#pragma unroll
  for (int r = 0; r < 8; ++r) {
    int row = r + halfg * 8;
    int q = qt * 128 + wid * 16 + row;
    float inv = 1.0f / lrow[r];
    size_t base = ((size_t)b * ATT_N + q) * ATT_C + h * ATT_D;
    AO[base + l16]      = f2bf(O0[r] * inv);
    AO[base + 16 + l16] = f2bf(O1[r] * inv);
  }
}

// ------------------------------- launcher ----------------------------------
extern "C" void kernel_launch(void* const* d_in, const int* in_sizes, int n_in,
                              void* d_out, int out_size, void* d_ws,
                              size_t ws_size, hipStream_t stream) {
  const float* x      = (const float*)d_in[0];  // [4,1024,768]
  const float* w_qkv  = (const float*)d_in[1];  // [768,2304]
  const float* w_proj = (const float*)d_in[2];  // [768,768]
  float* out = (float*)d_out;                   // [4096,768] f32

  const size_t XN  = (size_t)ATT_B * ATT_N * ATT_C;   // 3145728
  const size_t WQN = (size_t)ATT_C * QKV_N;           // 1769472
  const size_t WPN = (size_t)ATT_C * ATT_C;           // 589824
  const size_t HN  = (size_t)ATT_BH * ATT_N * ATT_D;  // 3145728 per Q/K/V

  unsigned short* xb  = (unsigned short*)d_ws;
  unsigned short* wqb = xb + XN;
  unsigned short* wpb = wqb + WQN;
  unsigned short* q   = wpb + WPN;   // qkv contiguous: [s][bh][n][d]
  unsigned short* k   = q + HN;
  unsigned short* v   = k + HN;
  unsigned short* ao  = v + HN;      // [4096][768] bf16

  cvt_bf16_kernel<<<(XN + 255) / 256, 256, 0, stream>>>(x, xb, (int)XN);
  cvt_bf16_kernel<<<(WQN + 255) / 256, 256, 0, stream>>>(w_qkv, wqb, (int)WQN);
  cvt_bf16_kernel<<<(WPN + 255) / 256, 256, 0, stream>>>(w_proj, wpb, (int)WPN);

  // QKV projection: [4096,768] x [768,2304] -> scattered bf16 Q/K/V
  dim3 g1(ATT_B * ATT_N / GTM, QKV_N / GTN);
  gemm_wmma_bf16_kernel<<<g1, 256, 0, stream>>>(xb, wqb, nullptr, q,
                                                ATT_B * ATT_N, QKV_N, ATT_C, 1);
  // attention
  dim3 g2(ATT_BH, ATT_N / 128);
  attn_wmma_kernel<<<g2, 256, 0, stream>>>(q, k, v, ao);

  // output projection: [4096,768] x [768,768] -> f32 out
  dim3 g3(ATT_B * ATT_N / GTM, ATT_C / GTN);
  gemm_wmma_bf16_kernel<<<g3, 256, 0, stream>>>(ao, wpb, out, nullptr,
                                                ATT_B * ATT_N, ATT_C, ATT_C, 0);
}
